// GraphAttentionLayer_2413771620693
// MI455X (gfx1250) — matrix operand order repaired and checked by reading, not compiled, not run on hardware
//
#include <hip/hip_runtime.h>

// MI455X / gfx1250, wave32. Fused flash-style GAT layer:
//   k_wh     : Wh = h @ W                         (v_wmma_f32_16x16x32_f16)
//   k_scores : s_src = Wh@a[:64], s_dst = Wh@a[64:]
//   k_pack   : Wh -> f16 dword-packed in WMMA B-operand layout
//   k_attn   : single pass over A (256 MiB @ 23.3 TB/s ~= 11.5 us floor):
//              online softmax (per-wave K partition) fused with P@Wh WMMA,
//              A streamed via global_load_async_to_lds_b128 double buffering
//              (inline asm; ASYNCcnt-tracked, s_wait_asynccnt for sync).

typedef __attribute__((ext_vector_type(16))) _Float16 v16h;
typedef __attribute__((ext_vector_type(8)))  float    v8f;

#define N_NODES   8192
#define IN_DIM    256
#define D_OUT     64
#define ALPHA_NS  0.2f
#define MASK_FILL 1e-15f

// ---- gfx1250 async-to-LDS path (feature probe via the builtin's presence;
//      emission via inline asm so we control operand types ourselves) -------
#if defined(__has_builtin)
#  if __has_builtin(__builtin_amdgcn_global_load_async_to_lds_b128)
#    define HAVE_ASYNC_LDS 1
#  endif
#endif
#ifndef HAVE_ASYNC_LDS
#  define HAVE_ASYNC_LDS 0
#endif

#if HAVE_ASYNC_LDS
// per ISA: flat aperture puts the LDS byte offset in the low 32 address bits
__device__ __forceinline__ unsigned lds_offset(const void* p)
{
    return (unsigned)(unsigned long long)p;
}
// GLOBAL_LOAD_ASYNC_TO_LDS_B128: VDST = LDS address VGPR, VADDR = 64-bit addr
__device__ __forceinline__ void async_load_b128(const void* gptr, unsigned loff)
{
    asm volatile("global_load_async_to_lds_b128 %0, %1, off"
                 :: "v"(loff), "v"((unsigned long long)gptr)
                 : "memory");
}
#  if __has_builtin(__builtin_amdgcn_s_wait_asynccnt)
#    define WAIT_ASYNCCNT(n) __builtin_amdgcn_s_wait_asynccnt(n)
#  else
#    define WAIT_ASYNCCNT(n) asm volatile("s_wait_asynccnt %0" ::"i"(n) : "memory")
#  endif
#endif

// ---------------------------------------------------------------------------
// Kernel 1: Wh = h @ W.  Block = 16 rows, 4 waves = 4 col tiles.
// A-operand (16x32 f16): lane l: M=l&15; lanes 0-15 K{0-7,16-23}, 16-31 K{8-15,24-31}.
// B-operand (32x16 f16): lane n: N=n&15; lanes 0-15 K{0-7,16-23}, 16-31 K{8-15,24-31}.
// ---------------------------------------------------------------------------
__global__ __launch_bounds__(128)
void k_wh(const float* __restrict__ h, const float* __restrict__ W,
          float* __restrict__ Wh)
{
    const int lane    = threadIdx.x & 31;
    const int wave    = threadIdx.x >> 5;
    const int rowbase = blockIdx.x * 16;
    const int lr      = lane & 15;
    const int hi      = lane >> 4;
    const int g       = hi * 8;
    const int col     = wave * 16 + lr;

    v8f acc = {};
    for (int kk = 0; kk < IN_DIM; kk += 32) {
        const float* hrow = h + (size_t)(rowbase + lr) * IN_DIM + kk + g;
        v16h a;
#pragma unroll
        for (int e = 0; e < 8; ++e)  a[e] = (_Float16)hrow[e];
#pragma unroll
        for (int e = 8; e < 16; ++e) a[e] = (_Float16)hrow[e + 8];

        const float* wcol = W + (size_t)(kk + g) * D_OUT + col;
        v16h b;
#pragma unroll
        for (int e = 0; e < 16; ++e) b[e] = (_Float16)wcol[(size_t)(e < 8 ? e : e + 8) * D_OUT];

        acc = __builtin_amdgcn_wmma_f32_16x16x32_f16(false, a, false, b,
                                                     (short)0, acc, false, false);
    }
#pragma unroll
    for (int k = 0; k < 8; ++k)
        Wh[(size_t)(rowbase + k + hi * 8) * D_OUT + col] = acc[k];
}

// ---------------------------------------------------------------------------
// Kernel 2: s_src / s_dst row dot products (one wave per row).
// ---------------------------------------------------------------------------
__global__ __launch_bounds__(256)
void k_scores(const float* __restrict__ Wh, const float* __restrict__ avec,
              float* __restrict__ s_src, float* __restrict__ s_dst)
{
    const int lane = threadIdx.x & 31;
    const int row  = blockIdx.x * 8 + (threadIdx.x >> 5);
    const float x0 = Wh[(size_t)row * D_OUT + lane];
    const float x1 = Wh[(size_t)row * D_OUT + 32 + lane];
    float ss = x0 * avec[lane]         + x1 * avec[32 + lane];
    float sd = x0 * avec[D_OUT + lane] + x1 * avec[D_OUT + 32 + lane];
#pragma unroll
    for (int off = 16; off > 0; off >>= 1) {
        ss += __shfl_xor(ss, off, 32);
        sd += __shfl_xor(sd, off, 32);
    }
    if (lane == 0) { s_src[row] = ss; s_dst[row] = sd; }
}

// ---------------------------------------------------------------------------
// Kernel 3: pack Wh into WMMA B-operand order (f16, dword-packed) so k_attn
// loads each 32x16 B tile as two contiguous b128 per lane.
// ---------------------------------------------------------------------------
__global__ __launch_bounds__(256)
void k_pack(const float* __restrict__ Wh, unsigned* __restrict__ WhB)
{
    const int tid  = blockIdx.x * 256 + threadIdx.x;    // N*D/2 dwords
    const int v    = tid & 7;
    const int lane = (tid >> 3) & 31;
    const int t    = (tid >> 8) & 3;
    const int c    = tid >> 10;
    const int col  = t * 16 + (lane & 15);
    const int k0   = c * 32 + (lane >> 4) * 8 + (v >> 2) * 16 + 2 * (v & 3);
    const _Float16 lo = (_Float16)Wh[(size_t)k0 * D_OUT + col];
    const _Float16 hi = (_Float16)Wh[(size_t)(k0 + 1) * D_OUT + col];
    WhB[tid] = (unsigned)__builtin_bit_cast(unsigned short, lo)
             | ((unsigned)__builtin_bit_cast(unsigned short, hi) << 16);
}

// ---------------------------------------------------------------------------
// Kernel 4 (fused): one streaming pass over A.
// Block = 16 output rows, 4 waves partition the 8192 neighbors in 32-chunks.
// Per wave: online (m,l) per row + 4 f32 WMMA accumulator tiles, rescaled by
// exp(m_old-m_new) via lane shuffles only when the row max actually moves.
// Wave partials merged through LDS with the standard log-sum-exp combine.
// ---------------------------------------------------------------------------
__global__ __launch_bounds__(128)
void k_attn(const float* __restrict__ A, const unsigned* __restrict__ WhB,
            const float* __restrict__ s_src, const float* __restrict__ s_dst,
            float* __restrict__ out)
{
    __shared__ float red[4][16][64];
    __shared__ float mbuf[4][16];
    __shared__ float lbuf[4][16];
#if HAVE_ASYNC_LDS
    __shared__ float sbuf[4][2][16][32];     // per-wave double-buffered A tile
#endif
    const int lane    = threadIdx.x & 31;
    const int wave    = threadIdx.x >> 5;
    const int rowbase = blockIdx.x * 16;
    const int lr      = lane & 15;
    const int hi      = lane >> 4;
    const int g       = hi * 8;               // K-group offset, also hi*8
    const int row     = rowbase + lr;
    const float si    = s_src[row];

    v8f acc0 = {}, acc1 = {}, acc2 = {}, acc3 = {};
    float m = -__builtin_inff(), l = 0.f;

    const int NI = (N_NODES / 32) / 4;         // 64 chunks per wave

#if HAVE_ASYNC_LDS
    const int sub = lane >> 3;                 // row group within chunk
    const int seg = lane & 7;                  // 16B segment within row
    {   // prologue: stage this wave's chunk 0
        const int jb = wave * 32;
#pragma unroll
        for (int q = 0; q < 4; ++q) {
            const int r4 = q * 4 + sub;
            async_load_b128(A + (size_t)(rowbase + r4) * N_NODES + jb + seg * 4,
                            lds_offset(&sbuf[wave][0][r4][seg * 4]));
        }
    }
#endif

    for (int i = 0; i < NI; ++i) {
        const int c     = wave + i * 4;
        const int jbase = c * 32;

        float av[16];
#if HAVE_ASYNC_LDS
        const int cur = i & 1;
        if (i + 1 < NI) {                      // stage next chunk, wait prev
            const int jn = (c + 4) * 32;
#pragma unroll
            for (int q = 0; q < 4; ++q) {
                const int r4 = q * 4 + sub;
                async_load_b128(A + (size_t)(rowbase + r4) * N_NODES + jn + seg * 4,
                                lds_offset(&sbuf[wave][cur ^ 1][r4][seg * 4]));
            }
            WAIT_ASYNCCNT(4);
        } else {
            WAIT_ASYNCCNT(0);
        }
        {
            const float4* p0 = (const float4*)&sbuf[wave][cur][lr][g];
            const float4* p1 = (const float4*)&sbuf[wave][cur][lr][g + 16];
            const float4 q0 = p0[0], q1 = p0[1], q2 = p1[0], q3 = p1[1];
            av[0]=q0.x;  av[1]=q0.y;  av[2]=q0.z;  av[3]=q0.w;
            av[4]=q1.x;  av[5]=q1.y;  av[6]=q1.z;  av[7]=q1.w;
            av[8]=q2.x;  av[9]=q2.y;  av[10]=q2.z; av[11]=q2.w;
            av[12]=q3.x; av[13]=q3.y; av[14]=q3.z; av[15]=q3.w;
        }
#else
        {
            const float* Ar = A + (size_t)row * N_NODES;
            const float4* p0 = (const float4*)(Ar + jbase + g);
            const float4* p1 = (const float4*)(Ar + jbase + g + 16);
            const float4 q0 = p0[0], q1 = p0[1], q2 = p1[0], q3 = p1[1];
            av[0]=q0.x;  av[1]=q0.y;  av[2]=q0.z;  av[3]=q0.w;
            av[4]=q1.x;  av[5]=q1.y;  av[6]=q1.z;  av[7]=q1.w;
            av[8]=q2.x;  av[9]=q2.y;  av[10]=q2.z; av[11]=q2.w;
            av[12]=q3.x; av[13]=q3.y; av[14]=q3.z; av[15]=q3.w;
        }
#endif
        // destination scores for this lane's 16 K-slots
        float sd[16];
        {
            const float4 s0 = *(const float4*)(s_dst + jbase + g);
            const float4 s1 = *(const float4*)(s_dst + jbase + g + 4);
            const float4 s2 = *(const float4*)(s_dst + jbase + g + 16);
            const float4 s3 = *(const float4*)(s_dst + jbase + g + 20);
            sd[0]=s0.x;  sd[1]=s0.y;  sd[2]=s0.z;  sd[3]=s0.w;
            sd[4]=s1.x;  sd[5]=s1.y;  sd[6]=s1.z;  sd[7]=s1.w;
            sd[8]=s2.x;  sd[9]=s2.y;  sd[10]=s2.z; sd[11]=s2.w;
            sd[12]=s3.x; sd[13]=s3.y; sd[14]=s3.z; sd[15]=s3.w;
        }

        // masked leaky scores + online max
        float val[16];
        float vmax = -__builtin_inff();
#pragma unroll
        for (int e = 0; e < 16; ++e) {
            const float x  = si + sd[e];
            const float lk = x > 0.f ? x : ALPHA_NS * x;
            const float v  = av[e] > 0.f ? lk : MASK_FILL;
            val[e] = v;
            vmax = fmaxf(vmax, v);
        }
        vmax = fmaxf(vmax, __shfl_xor(vmax, 16, 32));     // both row halves
        const float mnew  = fmaxf(m, vmax);
        const float scale = __expf(m - mnew);             // 0 on first chunk
        m = mnew;

        v16h pa;
        float ls = 0.f;
#pragma unroll
        for (int e = 0; e < 16; ++e) {
            const float p = __expf(val[e] - mnew);
            pa[e] = (_Float16)p;
            ls += p;
        }
        ls += __shfl_xor(ls, 16, 32);
        l = l * scale + ls;

        // rescale accumulators only when some row max moved (wave-uniform test)
        if (!__all(scale == 1.0f)) {
#pragma unroll
            for (int k = 0; k < 8; ++k) {
                const float sk = __shfl(scale, k + g, 32); // row k+hi*8 factor
                acc0[k] *= sk; acc1[k] *= sk; acc2[k] *= sk; acc3[k] *= sk;
            }
        }

        const unsigned* wb = WhB + (size_t)c * 1024 + (size_t)lane * 8;
        const v16h b0 = *(const v16h*)(wb);
        const v16h b1 = *(const v16h*)(wb + 256);
        const v16h b2 = *(const v16h*)(wb + 512);
        const v16h b3 = *(const v16h*)(wb + 768);

        acc0 = __builtin_amdgcn_wmma_f32_16x16x32_f16(false, pa, false, b0, (short)0, acc0, false, false);
        acc1 = __builtin_amdgcn_wmma_f32_16x16x32_f16(false, pa, false, b1, (short)0, acc1, false, false);
        acc2 = __builtin_amdgcn_wmma_f32_16x16x32_f16(false, pa, false, b2, (short)0, acc2, false, false);
        acc3 = __builtin_amdgcn_wmma_f32_16x16x32_f16(false, pa, false, b3, (short)0, acc3, false, false);
    }

    // ---- cross-wave merge: log-sum-exp combine of (m, l, acc) -------------
    if (hi == 0) { mbuf[wave][lr] = m; lbuf[wave][lr] = l; }
#pragma unroll
    for (int k = 0; k < 8; ++k) {
        const int r = k + g;                   // k + hi*8
        red[wave][r][ 0 + lr] = acc0[k];
        red[wave][r][16 + lr] = acc1[k];
        red[wave][r][32 + lr] = acc2[k];
        red[wave][r][48 + lr] = acc3[k];
    }
    __syncthreads();

    for (int o = threadIdx.x; o < 16 * 64; o += 128) {
        const int r = o >> 6, col = o & 63;
        const float M = fmaxf(fmaxf(mbuf[0][r], mbuf[1][r]),
                              fmaxf(mbuf[2][r], mbuf[3][r]));
        float num = 0.f, den = 0.f;
#pragma unroll
        for (int w = 0; w < 4; ++w) {
            const float f = __expf(mbuf[w][r] - M);
            num += f * red[w][r][col];
            den += f * lbuf[w][r];
        }
        const float hv = num / den;
        out[(size_t)(rowbase + r) * D_OUT + col] =
            hv > 0.f ? hv : __expf(hv) - 1.f;            // ELU (alpha=1)
    }
}

// ---------------------------------------------------------------------------
extern "C" void kernel_launch(void* const* d_in, const int* in_sizes, int n_in,
                              void* d_out, int out_size, void* d_ws, size_t ws_size,
                              hipStream_t stream)
{
    const float* h = (const float*)d_in[0];
    const float* A = (const float*)d_in[1];
    const float* W = (const float*)d_in[2];
    const float* a = (const float*)d_in[3];
    float* out = (float*)d_out;

    char* ws = (char*)d_ws;
    float*    Wh    = (float*)ws;    ws += (size_t)N_NODES * D_OUT * sizeof(float);        // 2 MiB
    unsigned* WhB   = (unsigned*)ws; ws += (size_t)N_NODES * D_OUT / 2 * sizeof(unsigned); // 1 MiB
    float*    s_src = (float*)ws;    ws += N_NODES * sizeof(float);
    float*    s_dst = (float*)ws;    ws += N_NODES * sizeof(float);

    k_wh    <<<N_NODES / 16, 128, 0, stream>>>(h, W, Wh);
    k_scores<<<N_NODES / 8,  256, 0, stream>>>(Wh, a, s_src, s_dst);
    k_pack  <<<(N_NODES * D_OUT / 2) / 256, 256, 0, stream>>>(Wh, WhB);
    k_attn  <<<N_NODES / 16, 128, 0, stream>>>(A, WhB, s_src, s_dst, out);
}
